// TimeSeriesAttentionModel1_50182397887199
// MI455X (gfx1250) — compile-verified, matrix operand order checked by reading, not run on hardware
//
#include <hip/hip_runtime.h>

// MI455X / gfx1250, wave32. Matrix math via V_WMMA_F32_16X16X32_BF16.
typedef __bf16 bf16_t;
typedef bf16_t v16bf __attribute__((ext_vector_type(16)));
typedef bf16_t v8bf  __attribute__((ext_vector_type(8)));
typedef float  v8f   __attribute__((ext_vector_type(8)));

#define SPITCH 260   // LDS row pitch in floats (256 + 4 pad, breaks 64-bank aliasing)

#define WMMA_BF16(A, B, C) \
    __builtin_amdgcn_wmma_f32_16x16x32_bf16(false, (A), false, (B), (short)0, (C), false, false)

// 16-bit A/B fragment K mapping (ISA 16x32 table):
//   half j in [0,8):  K = hi*8 + j
//   half j in [8,16): K = 16 + hi*8 + (j-8)
// i.e. two contiguous 8-element runs at offsets hi*8 and 16+hi*8.

__device__ __forceinline__ v16bf frag_from_f32(const float* rowk, int hi) {
    const float* p0 = rowk + hi * 8;
    const float* p1 = rowk + 16 + hi * 8;
    v16bf f;
#pragma unroll
    for (int j = 0; j < 8; ++j) {
        f[j]     = (bf16_t)p0[j];
        f[j + 8] = (bf16_t)p1[j];
    }
    return f;
}

__device__ __forceinline__ v16bf frag_from_bf16(const bf16_t* rowk, int hi) {
    v8bf a = *(const v8bf*)(rowk + hi * 8);
    v8bf b = *(const v8bf*)(rowk + 16 + hi * 8);
    v16bf f;
#pragma unroll
    for (int j = 0; j < 8; ++j) { f[j] = a[j]; f[j + 8] = b[j]; }
    return f;
}

// ---------------------------------------------------------------------------
// Kernel 0: bulk f32 -> bf16 conversion (x and W into workspace), float4 granularity.
// ---------------------------------------------------------------------------
__global__ void cvt_bf16_kernel(const float* __restrict__ src,
                                bf16_t* __restrict__ dst, int n4) {
    const int i = blockIdx.x * blockDim.x + threadIdx.x;
    if (i < n4) {
        float4 v = *(const float4*)(src + i * 4);
        bf16_t* p = dst + i * 4;
        p[0] = (bf16_t)v.x; p[1] = (bf16_t)v.y;
        p[2] = (bf16_t)v.z; p[3] = (bf16_t)v.w;
    }
}

// ---------------------------------------------------------------------------
// Kernel 1: qkv projection, M=32768 (B*D*S), K=512 (L), N=1536 (3L).
//   q,k (f<1024)  -> qk[m][f]           (stride 1024, bf16)
//   v  (f>=1024)  -> vT[head][l][t]     (transposed per head, bf16) so the
//                    attn@v GEMM gets contiguous B fragments.
//   Register blocking 1Mx4N per wave; double-buffered fragment loads so the
//   compiler can issue partial s_wait_loadcnt and overlap VMEM with WMMA.
// ---------------------------------------------------------------------------
__global__ void qkv_gemm_kernel(const bf16_t* __restrict__ xb,
                                const bf16_t* __restrict__ Wb,
                                const float* __restrict__ bias,
                                bf16_t* __restrict__ qk,
                                bf16_t* __restrict__ vT) {
    const int lane = threadIdx.x & 31;
    const int r    = lane & 15;
    const int hi   = lane >> 4;
    const int wid  = blockIdx.x * 8 + (threadIdx.x >> 5);   // 0 .. 49151
    const int mtile = wid / 24;                             // 2048 M tiles
    const int ng    = wid % 24;                             // 24 groups of 4 N tiles
    const int m0 = mtile * 16;
    const int f0 = ng * 64;

    const bf16_t* arow  = xb + (long)(m0 + r) * 512;
    const bf16_t* brow0 = Wb + (long)(f0 + r) * 512;        // N tiles 16*512 apart

    v8f acc0 = {}, acc1 = {}, acc2 = {}, acc3 = {};
    // double-buffer: preload k-step 0
    v16bf a  = frag_from_bf16(arow, hi);
    v16bf b0 = frag_from_bf16(brow0, hi);
    v16bf b1 = frag_from_bf16(brow0 + 16 * 512, hi);
    v16bf b2 = frag_from_bf16(brow0 + 32 * 512, hi);
    v16bf b3 = frag_from_bf16(brow0 + 48 * 512, hi);
    for (int kb = 0; kb < 512; kb += 32) {
        const int kn = (kb + 32) & 511;                     // wraps on last iter (dead)
        __builtin_prefetch(arow + kn + 128, 0, 1);          // global_prefetch_b8
        v16bf an  = frag_from_bf16(arow + kn, hi);
        v16bf b0n = frag_from_bf16(brow0 + kn, hi);
        v16bf b1n = frag_from_bf16(brow0 + 16 * 512 + kn, hi);
        v16bf b2n = frag_from_bf16(brow0 + 32 * 512 + kn, hi);
        v16bf b3n = frag_from_bf16(brow0 + 48 * 512 + kn, hi);
        acc0 = WMMA_BF16(a, b0, acc0);
        acc1 = WMMA_BF16(a, b1, acc1);
        acc2 = WMMA_BF16(a, b2, acc2);
        acc3 = WMMA_BF16(a, b3, acc3);
        a = an; b0 = b0n; b1 = b1n; b2 = b2n; b3 = b3n;
    }

    const int head = m0 >> 8;          // 256 rows per head
    const int trow = m0 & 255;         // row-in-head base
#pragma unroll
    for (int u = 0; u < 4; ++u) {
        const v8f acc = (u == 0) ? acc0 : (u == 1) ? acc1 : (u == 2) ? acc2 : acc3;
        const int fc = f0 + u * 16 + r;
        const float bv = bias[fc];
        if (f0 + u * 16 < 1024) {
            // q/k tile: D[m=i+hi*8][n=r] -> qk[m][fc]
#pragma unroll
            for (int i = 0; i < 8; ++i)
                qk[(long)(m0 + i + hi * 8) * 1024 + fc] = (bf16_t)(acc[i] + bv);
        } else {
            // v tile, stored transposed: vT[head][l = fc-1024][t = trow+i+hi*8]
            bf16_t* vrow = vT + ((long)head * 512 + (fc - 1024)) * 256 + trow + hi * 8;
#pragma unroll
            for (int i = 0; i < 8; ++i)
                vrow[i] = (bf16_t)(acc[i] + bv);
        }
    }
}

// ---------------------------------------------------------------------------
// Kernel 2: one block per head (b,d). 512 threads = 16 waves.
//   LDS: full 256x256 score/attn matrix (pitch SPITCH).
//   Phase 1: scores = (q k^T)/sqrt(512) -> LDS   (WMMA, 4x T blocking, dbl-buf)
//   Phase 2: row softmax, post-softmax tridiagonal mask, stream full
//            2048-wide rows of the block-diagonal attention matrix.
//   Phase 3: out = attn @ vT^T                   (WMMA, 4x L blocking, dbl-buf)
// ---------------------------------------------------------------------------
__global__ void attention_kernel(const bf16_t* __restrict__ qk,
                                 const bf16_t* __restrict__ vT,
                                 float* __restrict__ out,       // [16, 2048, 512]
                                 float* __restrict__ attn_out)  // [16, 2048, 2048]
{
    extern __shared__ float sc[];          // 256 * SPITCH floats
    const int head = blockIdx.x;           // b*8 + d
    const int b = head >> 3;
    const int d = head & 7;
    const long m0 = (long)head * 256;      // row base inside qk

    const int lane  = threadIdx.x & 31;
    const int r     = lane & 15;
    const int hi    = lane >> 4;
    const int w     = threadIdx.x >> 5;    // wave 0..15
    const int srow0 = w * 16;              // this wave's 16-row strip

    const float scale = 0.04419417382415922f;  // 1/sqrt(512)

    // ---- Phase 1: score strip [srow0, srow0+16) x 256 into LDS ----
    const bf16_t* qrow = qk + (m0 + srow0 + r) * 1024;             // q, K contiguous
    for (int tg = 0; tg < 4; ++tg) {                               // groups of 4 T tiles
        const int t0 = tg * 64;
        const bf16_t* krow0 = qk + (m0 + t0 + r) * 1024 + 512;     // k
        v8f acc0 = {}, acc1 = {}, acc2 = {}, acc3 = {};
        v16bf a  = frag_from_bf16(qrow, hi);
        v16bf b0 = frag_from_bf16(krow0, hi);
        v16bf b1 = frag_from_bf16(krow0 + 16 * 1024, hi);
        v16bf b2 = frag_from_bf16(krow0 + 32 * 1024, hi);
        v16bf b3 = frag_from_bf16(krow0 + 48 * 1024, hi);
        for (int kb = 0; kb < 512; kb += 32) {
            const int kn = (kb + 32) & 511;
            v16bf an  = frag_from_bf16(qrow + kn, hi);
            v16bf b0n = frag_from_bf16(krow0 + kn, hi);
            v16bf b1n = frag_from_bf16(krow0 + 16 * 1024 + kn, hi);
            v16bf b2n = frag_from_bf16(krow0 + 32 * 1024 + kn, hi);
            v16bf b3n = frag_from_bf16(krow0 + 48 * 1024 + kn, hi);
            acc0 = WMMA_BF16(a, b0, acc0);
            acc1 = WMMA_BF16(a, b1, acc1);
            acc2 = WMMA_BF16(a, b2, acc2);
            acc3 = WMMA_BF16(a, b3, acc3);
            a = an; b0 = b0n; b1 = b1n; b2 = b2n; b3 = b3n;
        }
#pragma unroll
        for (int u = 0; u < 4; ++u) {
            const v8f acc = (u == 0) ? acc0 : (u == 1) ? acc1 : (u == 2) ? acc2 : acc3;
#pragma unroll
            for (int i = 0; i < 8; ++i)
                sc[(srow0 + i + hi * 8) * SPITCH + t0 + u * 16 + r] = acc[i] * scale;
        }
    }
    __syncthreads();

    // ---- Phase 2a: per-row softmax + post-softmax mask (|s-t| <= 1 -> 0) ----
    if (threadIdx.x < 256) {
        const int s = threadIdx.x;
        float* row = &sc[s * SPITCH];
        float mx = -3.4e38f;
        for (int tt = 0; tt < 256; ++tt) { int t = (tt + s) & 255; mx = fmaxf(mx, row[t]); }
        float sum = 0.f;
        for (int tt = 0; tt < 256; ++tt) {
            int t = (tt + s) & 255;
            float e = __expf(row[t] - mx);
            row[t] = e;
            sum += e;
        }
        const float inv = 1.0f / sum;
        for (int tt = 0; tt < 256; ++tt) { int t = (tt + s) & 255; row[t] *= inv; }
        row[s] = 0.f;
        if (s > 0)   row[s - 1] = 0.f;
        if (s < 255) row[s + 1] = 0.f;
    }
    __syncthreads();

    // ---- Phase 2b: write block-diagonal attn matrix rows (zeros elsewhere) ----
    {
        const int s       = threadIdx.x >> 1;   // row in this head
        const int halfsel = threadIdx.x & 1;    // which 1024-col half
        const int cbase   = halfsel * 1024;
        const int blk0    = d * 256;            // this head's column block
        float* grow = attn_out + ((long)b * 2048 + d * 256 + s) * 2048 + cbase;
        for (int c4 = 0; c4 < 1024; c4 += 4) {
            const int c = cbase + c4;
            float4 vv = make_float4(0.f, 0.f, 0.f, 0.f);
            if (c >= blk0 && c < blk0 + 256) {
                const float* p = &sc[s * SPITCH + (c - blk0)];
                vv = make_float4(p[0], p[1], p[2], p[3]);
            }
            *(float4*)(grow + c4) = vv;
        }
    }
    __syncthreads();

    // ---- Phase 3: out[srow0..+16, :512] = attn @ v  (B from vT rows) ----
    const bf16_t* vTh = vT + (long)head * 512 * 256;
    for (int lg = 0; lg < 8; ++lg) {                 // groups of 4 L tiles
        const int l0 = lg * 64;
        const bf16_t* v0 = vTh + (long)(l0 + r) * 256;       // B column = vT row
        const bf16_t* v1 = vTh + (long)(l0 + 16 + r) * 256;
        const bf16_t* v2 = vTh + (long)(l0 + 32 + r) * 256;
        const bf16_t* v3 = vTh + (long)(l0 + 48 + r) * 256;
        v8f acc0 = {}, acc1 = {}, acc2 = {}, acc3 = {};
        v16bf b0 = frag_from_bf16(v0, hi);
        v16bf b1 = frag_from_bf16(v1, hi);
        v16bf b2 = frag_from_bf16(v2, hi);
        v16bf b3 = frag_from_bf16(v3, hi);
        for (int tb = 0; tb < 256; tb += 32) {
            const int tn = (tb + 32) & 255;
            v16bf b0n = frag_from_bf16(v0 + tn, hi);
            v16bf b1n = frag_from_bf16(v1 + tn, hi);
            v16bf b2n = frag_from_bf16(v2 + tn, hi);
            v16bf b3n = frag_from_bf16(v3 + tn, hi);
            // A: attn[srow0 + r][tb + K(j)] from LDS, f32 -> bf16 (shared by 4 tiles)
            v16bf a = frag_from_f32(&sc[(srow0 + r) * SPITCH + tb], hi);
            acc0 = WMMA_BF16(a, b0, acc0);
            acc1 = WMMA_BF16(a, b1, acc1);
            acc2 = WMMA_BF16(a, b2, acc2);
            acc3 = WMMA_BF16(a, b3, acc3);
            b0 = b0n; b1 = b1n; b2 = b2n; b3 = b3n;
        }
#pragma unroll
        for (int u = 0; u < 4; ++u) {
            const v8f acc = (u == 0) ? acc0 : (u == 1) ? acc1 : (u == 2) ? acc2 : acc3;
#pragma unroll
            for (int i = 0; i < 8; ++i)
                out[((long)b * 2048 + d * 256 + srow0 + i + hi * 8) * 512 + l0 + u * 16 + r] = acc[i];
        }
    }
}

// ---------------------------------------------------------------------------
extern "C" void kernel_launch(void* const* d_in, const int* in_sizes, int n_in,
                              void* d_out, int out_size, void* d_ws, size_t ws_size,
                              hipStream_t stream) {
    const float* x    = (const float*)d_in[0];   // [16,8,256,512]
    const float* W    = (const float*)d_in[1];   // [1536,512]
    const float* bias = (const float*)d_in[2];   // [1536]
    // d_in[3] = n (=256): mask is statically the tridiagonal band for these dims.

    const long NX  = 16L * 8 * 256 * 512;        // 16,777,216
    const long NW  = 1536L * 512;                //    786,432
    const long NQK = 32768L * 1024;              // 33,554,432
    const long NVT = 128L * 512 * 256;           // 16,777,216

    bf16_t* xb = (bf16_t*)d_ws;                  // x in bf16             (32 MB)
    bf16_t* Wb = xb + NX;                        // W in bf16             (1.5 MB)
    bf16_t* qk = Wb + NW;                        // [32768, 1024] q|k     (64 MB)
    bf16_t* vT = qk + NQK;                       // [128][512][256] v^T   (32 MB)

    float* out  = (float*)d_out;                 // output: 16*2048*512
    float* attn = out + (long)16 * 2048 * 512;   // attn matrix: 16*2048*2048

    // Kernel 0: one-pass f32 -> bf16 of x and W
    cvt_bf16_kernel<<<(int)(NX / 4 / 256), 256, 0, stream>>>(x, xb, (int)(NX / 4));
    cvt_bf16_kernel<<<(int)(NW / 4 / 256), 256, 0, stream>>>(W, Wb, (int)(NW / 4));

    // Kernel 1: 49152 wave-tiles (2048 M x 24 N-groups) / 8 waves per block
    qkv_gemm_kernel<<<6144, 256, 0, stream>>>(xb, Wb, bias, qk, vT);

    // Kernel 2: one block per (b,d) head; 260 KB dynamic LDS (<= 320 KB WGP budget)
    attention_kernel<<<128, 512, 256 * SPITCH * sizeof(float), stream>>>(qk, vT, out, attn);
}
